// MyGNN_OneStep_38139309589236
// MI455X (gfx1250) — matrix operand order repaired and checked by reading, not compiled, not run on hardware
//
#include <hip/hip_runtime.h>
#include <cstddef>

typedef __bf16 v16bf __attribute__((ext_vector_type(16)));
typedef __bf16 v8bf  __attribute__((ext_vector_type(8)));
typedef __bf16 v4bf  __attribute__((ext_vector_type(4)));
typedef float  v8f   __attribute__((ext_vector_type(8)));

// ---------------------------------------------------------------------------
// Fragment helpers (this file's contiguous k order k = 16*half + s, 16-bit A 16x32 / B 32x16)
// ---------------------------------------------------------------------------
// A: lane<16 -> row=lane, K={0..15}; lane>=16 -> row=lane-16, K={16..31}
// caller passes p = &row[k0 + (lane>>4)*16]; we read p[0..7] and p[8..15].
__device__ __forceinline__ v16bf ld_frag_a(const __bf16* p) {
  v8bf lo = *reinterpret_cast<const v8bf*>(p);
  v8bf hi = *reinterpret_cast<const v8bf*>(p + 8);
  return __builtin_shufflevector(lo, hi, 0,1,2,3,4,5,6,7,8,9,10,11,12,13,14,15);
}

__device__ __forceinline__ v8f wmma_bf16(v16bf a, v16bf b, v8f c) {
  return __builtin_amdgcn_wmma_f32_16x16x32_bf16(false, a, false, b, (short)0, c,
                                                 false, false);
}

__device__ __forceinline__ float silu(float y) { return y / (1.0f + __expf(-y)); }

// ---------------------------------------------------------------------------
// Prep kernels: zero accumulators; transpose f32 weights -> bf16 Wt[n][k]
// ---------------------------------------------------------------------------
__global__ void zero_f32(float* p, long n) {
  long i = (long)blockIdx.x * blockDim.x + threadIdx.x;
  long stride = (long)gridDim.x * blockDim.x;
  for (; i < n; i += stride) p[i] = 0.0f;
}

__global__ void transpose_w_bf16(const float* __restrict__ src, __bf16* __restrict__ dst,
                                 int K, int N) {
  int idx = blockIdx.x * blockDim.x + threadIdx.x;   // over N*K, k fastest
  if (idx < N * K) {
    int n = idx / K;
    int k = idx - n * K;
    dst[(long)n * K + k] = (__bf16)src[(long)k * N + n];   // Wt[n][k] = W[k][n]
  }
}

// ---------------------------------------------------------------------------
// Fused [Linear(KD->KD), LN, SiLU, Linear(KD->128), LN] + mode epilogue.
// MODE 0: edge   (gather e_h|h_src[s]|h_dst[d]; e_out=e_h+e_new; atomic scatter)
// MODE 1: dst    (gather h_dst|m_sum; out = h_dst + (deg>0 ? h_new : 0))
// MODE 2: src    (gather h_src;       out = h_src + h_new)
// ---------------------------------------------------------------------------
template <int KD, int MODE>
__global__ __launch_bounds__(256) void fused_mlp(
    const float* __restrict__ xA, const float* __restrict__ xB,
    const float* __restrict__ xC,
    const int* __restrict__ src_idx, const int* __restrict__ dst_idx,
    const __bf16* __restrict__ w0t, const float* __restrict__ b0,
    const float* __restrict__ g0, const float* __restrict__ bt0,
    const __bf16* __restrict__ w1t, const float* __restrict__ b1,
    const float* __restrict__ g1, const float* __restrict__ bt1,
    float* __restrict__ out, const float* __restrict__ resid,
    float* __restrict__ msum, float* __restrict__ deg, int M) {
  constexpr int TM = 128;           // rows per workgroup
  constexpr int D = 128;            // output width
  constexpr int SA = KD + 8;        // padded LDS pitch (x2B -> multiple of 16B)
  constexpr int CS = 132;           // f32 pitch for final C tile
  constexpr int NK = KD / 32;       // k-steps (A fragments held in registers)
  constexpr size_t RB = (size_t)TM * SA * 2;   // one bf16 tile region

  extern __shared__ char smem[];
  __bf16* Abuf = (__bf16*)smem;                              // GEMM0 A tile
  __bf16* Xbuf = (__bf16*)(smem + RB);                       // hidden activations
  float*  Cbuf = (float*)(smem + 2 * RB);                    // GEMM1 C tile (f32)
  float*  sMean = (float*)(smem + 2 * RB + (size_t)TM * CS * 4);
  float*  sRstd = sMean + TM;
  int*    sSrc  = (int*)(sRstd + TM);
  int*    sDst  = sSrc + TM;

  const int tid  = threadIdx.x;
  const int wv   = tid >> 5;
  const int lane = tid & 31;
  const int half = lane >> 4;
  const int ln   = lane & 15;
  const int base = blockIdx.x * TM;

  // ---- stage per-row gather indices (edge mode) ----
  if (MODE == 0 && tid < TM) {
    int gr = base + tid;
    sSrc[tid] = (gr < M) ? src_idx[gr] : 0;
    sDst[tid] = (gr < M) ? dst_idx[gr] : 0;
  }
  if (MODE == 0) __syncthreads();

  // ---- gather + f32->bf16 convert into Abuf (4 elements / iteration) ----
  for (int q = tid; q < TM * (KD / 4); q += 256) {
    int row = q / (KD / 4);
    int c4 = (q - row * (KD / 4)) * 4;
    int gr = base + row;
    float4 v = make_float4(0.f, 0.f, 0.f, 0.f);
    if (gr < M) {
      if (MODE == 0) {
        if (c4 < 128)      v = *(const float4*)&xA[(long)gr * 128 + c4];
        else if (c4 < 256) v = *(const float4*)&xB[(long)sSrc[row] * 128 + (c4 - 128)];
        else               v = *(const float4*)&xC[(long)sDst[row] * 128 + (c4 - 256)];
      } else if (MODE == 1) {
        v = (c4 < 128) ? *(const float4*)&xA[(long)gr * 128 + c4]
                       : *(const float4*)&xB[(long)gr * 128 + (c4 - 128)];
      } else {
        v = *(const float4*)&xA[(long)gr * 128 + c4];
      }
    }
    v4bf pk = { (__bf16)v.x, (__bf16)v.y, (__bf16)v.z, (__bf16)v.w };
    *(v4bf*)&Abuf[row * SA + c4] = pk;
  }
  __syncthreads();

  // ---- GEMM0: X1 = A(128xKD) * W0t^T + b0 ; per-wave 16-row strip ----
  {
    const __bf16* arow = Abuf + (wv * 16 + ln) * SA + half * 16;
    v16bf afr[NK];
#pragma unroll
    for (int kt = 0; kt < NK; ++kt) afr[kt] = ld_frag_a(arow + kt * 32);

    float s[8], s2[8];
#pragma unroll
    for (int j = 0; j < 8; ++j) { s[j] = 0.0f; s2[j] = 0.0f; }
    for (int nc = 0; nc < KD / 16; ++nc) {
      int col = nc * 16 + ln;
      float bias = b0[col];
      v8f acc;
#pragma unroll
      for (int j = 0; j < 8; ++j) acc[j] = bias;
      const __bf16* brow = w0t + (long)col * KD + half * 16;
#pragma unroll
      for (int kt = 0; kt < NK; ++kt) {
        v16bf b = *reinterpret_cast<const v16bf*>(brow + kt * 32);
        acc = wmma_bf16(afr[kt], b, acc);
      }
#pragma unroll
      for (int j = 0; j < 8; ++j) {
        float v = acc[j];
        s[j] += v; s2[j] += v * v;
        int row = wv * 16 + (half ? j + 8 : j);
        Xbuf[row * SA + col] = (__bf16)v;
      }
    }
    // butterfly reduce LN stats within each 16-lane half
#pragma unroll
    for (int j = 0; j < 8; ++j) {
#pragma unroll
      for (int m = 8; m >= 1; m >>= 1) {
        s[j]  += __shfl_xor(s[j],  m, 32);
        s2[j] += __shfl_xor(s2[j], m, 32);
      }
      float mean = s[j] * (1.0f / KD);
      float var  = s2[j] * (1.0f / KD) - mean * mean;
      float rstd = rsqrtf(var + 1e-5f);
      int row = wv * 16 + (half ? j + 8 : j);
      if (ln == 0) { sMean[row] = mean; sRstd[row] = rstd; }
    }
  }
  __syncthreads();

  // ---- LN + SiLU applied in-place on Xbuf (4 elements / iteration) ----
  for (int q = tid; q < TM * (KD / 4); q += 256) {
    int row = q / (KD / 4);
    int c4 = (q - row * (KD / 4)) * 4;
    float mean = sMean[row], rstd = sRstd[row];
    v4bf x = *(v4bf*)&Xbuf[row * SA + c4];
    float4 g = *(const float4*)&g0[c4];
    float4 bt = *(const float4*)&bt0[c4];
    v4bf r;
    r[0] = (__bf16)silu(((float)x[0] - mean) * rstd * g.x + bt.x);
    r[1] = (__bf16)silu(((float)x[1] - mean) * rstd * g.y + bt.y);
    r[2] = (__bf16)silu(((float)x[2] - mean) * rstd * g.z + bt.z);
    r[3] = (__bf16)silu(((float)x[3] - mean) * rstd * g.w + bt.w);
    *(v4bf*)&Xbuf[row * SA + c4] = r;
  }
  __syncthreads();

  // ---- GEMM1: C = X1(128xKD) * W1t^T + b1 -> Cbuf (f32) ----
  {
    const __bf16* arow = Xbuf + (wv * 16 + ln) * SA + half * 16;
    v16bf afr[NK];
#pragma unroll
    for (int kt = 0; kt < NK; ++kt) afr[kt] = ld_frag_a(arow + kt * 32);

    float s[8], s2[8];
#pragma unroll
    for (int j = 0; j < 8; ++j) { s[j] = 0.0f; s2[j] = 0.0f; }
    for (int nc = 0; nc < D / 16; ++nc) {
      int col = nc * 16 + ln;
      float bias = b1[col];
      v8f acc;
#pragma unroll
      for (int j = 0; j < 8; ++j) acc[j] = bias;
      const __bf16* brow = w1t + (long)col * KD + half * 16;
#pragma unroll
      for (int kt = 0; kt < NK; ++kt) {
        v16bf b = *reinterpret_cast<const v16bf*>(brow + kt * 32);
        acc = wmma_bf16(afr[kt], b, acc);
      }
#pragma unroll
      for (int j = 0; j < 8; ++j) {
        float v = acc[j];
        s[j] += v; s2[j] += v * v;
        int row = wv * 16 + (half ? j + 8 : j);
        Cbuf[row * CS + col] = v;
      }
    }
#pragma unroll
    for (int j = 0; j < 8; ++j) {
#pragma unroll
      for (int m = 8; m >= 1; m >>= 1) {
        s[j]  += __shfl_xor(s[j],  m, 32);
        s2[j] += __shfl_xor(s2[j], m, 32);
      }
      float mean = s[j] * (1.0f / D);
      float var  = s2[j] * (1.0f / D) - mean * mean;
      float rstd = rsqrtf(var + 1e-5f);
      int row = wv * 16 + (half ? j + 8 : j);
      if (ln == 0) { sMean[row] = mean; sRstd[row] = rstd; }
    }
  }
  __syncthreads();

  // ---- Epilogue: LN2, residual, (edge: scatter to m_sum/deg) ----
  for (int q = tid; q < TM * 32; q += 256) {
    int row = q >> 5;
    int c4 = (q & 31) << 2;
    int gr = base + row;
    if (gr >= M) continue;
    float mean = sMean[row], rstd = sRstd[row];
    const float* cp = &Cbuf[row * CS + c4];
    float4 g = *(const float4*)&g1[c4];
    float4 bt = *(const float4*)&bt1[c4];
    float4 v;
    v.x = (cp[0] - mean) * rstd * g.x + bt.x;
    v.y = (cp[1] - mean) * rstd * g.y + bt.y;
    v.z = (cp[2] - mean) * rstd * g.z + bt.z;
    v.w = (cp[3] - mean) * rstd * g.w + bt.w;
    long o = (long)gr * 128 + c4;
    float4 r = *(const float4*)&resid[o];
    if (MODE == 0) {
      float4 ov = make_float4(r.x + v.x, r.y + v.y, r.z + v.z, r.w + v.w);
      *(float4*)&out[o] = ov;
      int d = sDst[row];
      float* mp = &msum[(long)d * 128 + c4];
      atomicAdd(mp + 0, v.x);
      atomicAdd(mp + 1, v.y);
      atomicAdd(mp + 2, v.z);
      atomicAdd(mp + 3, v.w);
      if (c4 == 0) atomicAdd(&deg[d], 1.0f);
    } else if (MODE == 1) {
      float m = (deg[gr] > 0.0f) ? 1.0f : 0.0f;
      float4 ov = make_float4(r.x + m * v.x, r.y + m * v.y,
                              r.z + m * v.z, r.w + m * v.w);
      *(float4*)&out[o] = ov;
    } else {
      float4 ov = make_float4(r.x + v.x, r.y + v.y, r.z + v.z, r.w + v.w);
      *(float4*)&out[o] = ov;
    }
  }
}

// ---------------------------------------------------------------------------
// Host launcher
// ---------------------------------------------------------------------------
static inline size_t smem_bytes(int KD) {
  size_t RB = (size_t)(KD + 8) * 128 * 2;   // bf16 tile region
  return 2 * RB + (size_t)132 * 128 * 4 /*Cbuf*/
         + 128 * 4 * 2 /*mean,rstd*/ + 128 * 4 * 2 /*sidx,didx*/;
}

extern "C" void kernel_launch(void* const* d_in, const int* in_sizes, int n_in,
                              void* d_out, int out_size, void* d_ws, size_t ws_size,
                              hipStream_t stream) {
  const float* e_h   = (const float*)d_in[0];
  const float* h_src = (const float*)d_in[1];
  const float* h_dst = (const float*)d_in[2];
  const int* src_idx = (const int*)d_in[3];
  const int* dst_idx = (const int*)d_in[4];
  const float* ew0 = (const float*)d_in[5];  const float* eb0 = (const float*)d_in[6];
  const float* eg0 = (const float*)d_in[7];  const float* ebt0 = (const float*)d_in[8];
  const float* ew1 = (const float*)d_in[9];  const float* eb1 = (const float*)d_in[10];
  const float* eg1 = (const float*)d_in[11]; const float* ebt1 = (const float*)d_in[12];
  const float* dw0 = (const float*)d_in[13]; const float* db0 = (const float*)d_in[14];
  const float* dg0 = (const float*)d_in[15]; const float* dbt0 = (const float*)d_in[16];
  const float* dw1 = (const float*)d_in[17]; const float* db1 = (const float*)d_in[18];
  const float* dg1 = (const float*)d_in[19]; const float* dbt1 = (const float*)d_in[20];
  const float* sw0 = (const float*)d_in[21]; const float* sb0 = (const float*)d_in[22];
  const float* sg0 = (const float*)d_in[23]; const float* sbt0 = (const float*)d_in[24];
  const float* sw1 = (const float*)d_in[25]; const float* sb1 = (const float*)d_in[26];
  const float* sg1 = (const float*)d_in[27]; const float* sbt1 = (const float*)d_in[28];

  const int E  = in_sizes[0] / 128;
  const int NS = in_sizes[1] / 128;
  const int ND = in_sizes[2] / 128;

  // ---- workspace carve ----
  char* ws = (char*)d_ws;
  size_t off = 0;
  auto carve = [&](size_t bytes) -> char* {
    char* p = ws + off;
    off = (off + bytes + 255) & ~(size_t)255;
    return p;
  };
  float*  msum = (float*)carve((size_t)ND * 128 * 4);
  float*  deg  = (float*)carve((size_t)ND * 4);
  __bf16* w0t  = (__bf16*)carve((size_t)384 * 384 * 2);
  __bf16* w1t  = (__bf16*)carve((size_t)128 * 384 * 2);
  __bf16* dw0t = (__bf16*)carve((size_t)256 * 256 * 2);
  __bf16* dw1t = (__bf16*)carve((size_t)128 * 256 * 2);
  __bf16* sw0t = (__bf16*)carve((size_t)128 * 128 * 2);
  __bf16* sw1t = (__bf16*)carve((size_t)128 * 128 * 2);

  float* out = (float*)d_out;
  const long o_src = (long)E * 128;
  const long o_dst = o_src + (long)NS * 128;

  // ---- prep: zero accumulators, transpose weights to bf16 ----
  zero_f32<<<1024, 256, 0, stream>>>(msum, (long)ND * 128);
  zero_f32<<<64, 256, 0, stream>>>(deg, (long)ND);
  auto tr = [&](const float* s, __bf16* d, int K, int N) {
    int n = N * K;
    transpose_w_bf16<<<(n + 255) / 256, 256, 0, stream>>>(s, d, K, N);
  };
  tr(ew0, w0t, 384, 384);
  tr(ew1, w1t, 384, 128);
  tr(dw0, dw0t, 256, 256);
  tr(dw1, dw1t, 256, 128);
  tr(sw0, sw0t, 128, 128);
  tr(sw1, sw1t, 128, 128);

  // ---- src MLP (independent) ----
  fused_mlp<128, 2><<<(NS + 127) / 128, 256, smem_bytes(128), stream>>>(
      h_src, nullptr, nullptr, nullptr, nullptr,
      sw0t, sb0, sg0, sbt0, sw1t, sb1, sg1, sbt1,
      out + o_src, h_src, nullptr, nullptr, NS);

  // ---- edge MLP + scatter-sum (needs zeroed msum/deg; stream-ordered) ----
  fused_mlp<384, 0><<<(E + 127) / 128, 256, smem_bytes(384), stream>>>(
      e_h, h_src, h_dst, src_idx, dst_idx,
      w0t, eb0, eg0, ebt0, w1t, eb1, eg1, ebt1,
      out, e_h, msum, deg, E);

  // ---- dst MLP (consumes msum/deg) ----
  fused_mlp<256, 1><<<(ND + 127) / 128, 256, smem_bytes(256), stream>>>(
      h_dst, msum, nullptr, nullptr, nullptr,
      dw0t, db0, dg0, dbt0, dw1t, db1, dg1, dbt1,
      out + o_dst, h_dst, nullptr, deg, ND);
}
